// Model_17549236371459
// MI455X (gfx1250) — matrix operand order repaired and checked by reading, not compiled, not run on hardware
//
#include <hip/hip_runtime.h>

// ---------------------------------------------------------------------------
// TimeSeries LSTM enc-dec (B=128,S=512,E=7,D=1024,C=7,F=4,L=2,PRED=96) on
// MI455X / gfx1250.  Strategy:
//  * convert all big weights + activations to bf16, accumulate fp32
//  * batch the non-recurrent input-gate GEMMs (M=65536,N=4096,K=1024) as
//    WMMA bf16 GEMMs (v_wmma_f32_16x16x32_bf16)
//  * per-timestep recurrent GEMM (128x4096x1024) as a fused WMMA kernel whose
//    epilogue applies the LSTM cell (sigmoid/tanh) in-register; weights
//    (8.4MB bf16) stay L2-resident across the 512/96 sequential steps
//  * attention is a batched GEMV over the L2-resident bf16 enc_out
// ---------------------------------------------------------------------------

typedef __attribute__((ext_vector_type(16))) __bf16 v16bf;
typedef __attribute__((ext_vector_type(8)))  __bf16 v8bf;
typedef __attribute__((ext_vector_type(8)))  float  v8f;

union FragU { v16bf v; v8bf h[2]; };

// ---- WMMA fragment loaders (layouts per CDNA5 ISA §7.12.2) -----

// A fragment: 16x32 bf16 tile of row-major matrix (rows = M, ld in elements).
__device__ __forceinline__ v16bf load_fragA(const __bf16* __restrict__ base,
                                            int ld, int m0, int k0) {
    const int lane = threadIdx.x & 31;
    const __bf16* p = base + (size_t)(m0 + (lane & 15)) * ld
                           + (k0 + ((lane >> 4) << 3));   // +8 for hi half
    FragU u;
    u.h[0] = *(const v8bf*)p;          // K = base+0..7
    u.h[1] = *(const v8bf*)(p + 16);   // K = base+16..23
    return u.v;
}

// B fragment: 32x16 tile of B = W^T where W is row-major [N,K].
// lane = N column (= row n of W); K = base+0..7 and base+16..23, base +8 for lane>=16.
__device__ __forceinline__ v16bf load_fragB(const __bf16* __restrict__ W,
                                            int ld, int n0, int k0) {
    const int lane = threadIdx.x & 31;
    const __bf16* p = W + (size_t)(n0 + (lane & 15)) * ld
                        + (k0 + ((lane >> 4) << 3));      // +8 for hi half
    FragU u;
    u.h[0] = *(const v8bf*)p;          // K = base+0..7
    u.h[1] = *(const v8bf*)(p + 16);   // K = base+16..23
    return u.v;
}

__device__ __forceinline__ v8f zero_v8f() {
    v8f z = {0.f,0.f,0.f,0.f,0.f,0.f,0.f,0.f};
    return z;
}

__device__ __forceinline__ void store_tile_f32(float* __restrict__ C, int ldc,
                                               int m0, int n0, v8f acc) {
    const int lane = threadIdx.x & 31;
    const int col  = n0 + (lane & 15);
    const int r0   = m0 + ((lane >> 4) << 3);
#pragma unroll
    for (int v = 0; v < 8; ++v) C[(size_t)(r0 + v) * ldc + col] = acc[v];
}

__device__ __forceinline__ float sigmoidf_(float x) {
    return 1.0f / (1.0f + __expf(-x));
}

__device__ __forceinline__ float warp_red_sum(float v) {
#pragma unroll
    for (int o = 16; o > 0; o >>= 1) v += __shfl_xor(v, o, 32);
    return v;
}
__device__ __forceinline__ float warp_red_max(float v) {
#pragma unroll
    for (int o = 16; o > 0; o >>= 1) v = fmaxf(v, __shfl_xor(v, o, 32));
    return v;
}

// ---------------------------------------------------------------------------
// Utility kernels
// ---------------------------------------------------------------------------
__global__ void k_zero_u32(unsigned int* __restrict__ p, int n) {
    int i = blockIdx.x * blockDim.x + threadIdx.x;
    int st = gridDim.x * blockDim.x;
    for (; i < n; i += st) p[i] = 0u;
}

__global__ void k_f32_to_bf16(const float* __restrict__ src,
                              __bf16* __restrict__ dst, int n) {
    int i = blockIdx.x * blockDim.x + threadIdx.x;
    int st = gridDim.x * blockDim.x;
    for (; i < n; i += st) dst[i] = (__bf16)src[i];
}

// ---------------------------------------------------------------------------
// Embedding: circular conv1d(E=7,k=3) + time-feature linear -> bf16 [S,B,D]
// one block per (t,b); 256 threads cover D=1024
// ---------------------------------------------------------------------------
__global__ void k_embed(const float* __restrict__ x,      // [B,S,E]
                        const float* __restrict__ xm,     // [B,S,F]
                        const float* __restrict__ cw,     // [D,E,3]
                        const float* __restrict__ tw,     // [D,F]
                        __bf16* __restrict__ emb)         // [S,B,D]
{
    const int bid = blockIdx.x;          // t*128 + b
    const int t = bid >> 7, b = bid & 127;
    __shared__ float sx[21];
    __shared__ float sm[4];
    const int tid = threadIdx.x;
    if (tid < 21) {
        int j = tid / 7, e = tid % 7;
        int s2 = (t - 1 + j + 512) & 511;           // circular pad
        sx[tid] = x[((size_t)b * 512 + s2) * 7 + e];
    } else if (tid < 25) {
        sm[tid - 21] = xm[((size_t)b * 512 + t) * 4 + (tid - 21)];
    }
    __syncthreads();
    for (int d = tid; d < 1024; d += 256) {
        float acc = 0.f;
        const float* w = cw + (size_t)d * 21;
#pragma unroll
        for (int e = 0; e < 7; ++e)
#pragma unroll
            for (int j = 0; j < 3; ++j)
                acc += sx[j * 7 + e] * w[e * 3 + j];
        const float* tt = tw + (size_t)d * 4;
#pragma unroll
        for (int f = 0; f < 4; ++f) acc += sm[f] * tt[f];
        emb[((size_t)t * 128 + b) * 1024 + d] = (__bf16)acc;
    }
}

// ---------------------------------------------------------------------------
// Batched input-gate GEMM: G[M,4096] = X[M,1024] @ W[4096,1024]^T  (fp32 out)
// one wave computes a 16(M) x 64(N) strip (4 WMMA tiles, A-frag reused)
// ---------------------------------------------------------------------------
__global__ void k_gemm_gates(const __bf16* __restrict__ X,
                             const __bf16* __restrict__ W,
                             float* __restrict__ G, int Mtiles)
{
    const int wave = blockIdx.x * (blockDim.x >> 5) + (threadIdx.x >> 5);
    const int nt4 = wave & 63;          // 64 strips of 64 cols
    const int mt  = wave >> 6;
    if (mt >= Mtiles) return;
    const int m0 = mt << 4, n0 = nt4 << 6;

    v8f acc[4];
#pragma unroll
    for (int g = 0; g < 4; ++g) acc[g] = zero_v8f();

    for (int k0 = 0; k0 < 1024; k0 += 32) {
        v16bf a = load_fragA(X, 1024, m0, k0);
#pragma unroll
        for (int g = 0; g < 4; ++g) {
            v16bf bf = load_fragB(W, 1024, n0 + g * 16, k0);
            acc[g] = __builtin_amdgcn_wmma_f32_16x16x32_bf16(
                false, a, false, bf, (short)0, acc[g], false, false);
        }
    }
#pragma unroll
    for (int g = 0; g < 4; ++g)
        store_tile_f32(G, 4096, m0, n0 + g * 16, acc[g]);
}

// ---------------------------------------------------------------------------
// Fused LSTM step:
//   gates = [Xin @ Wih^T] + Hprev @ Whh^T + Gpre + bias ; apply cell
// Each wave owns one 16(batch) x 16(hidden) tile and its 4 gate tiles
// (i,f,g,o live at n, n+1024, n+2048, n+3072) so the nonlinearity fuses.
// grid = 128 blocks x 128 threads (512 waves = 8 m-tiles * 64 d-tiles).
// ---------------------------------------------------------------------------
__global__ void k_lstm_step(const __bf16* __restrict__ Xin,   // [128,1024] or null
                            const __bf16* __restrict__ Wih,   // [4096,1024]
                            const __bf16* __restrict__ Hprev, // [128,1024]
                            const __bf16* __restrict__ Whh,   // [4096,1024]
                            const float*  __restrict__ Gpre,  // [128,4096] or null
                            const float*  __restrict__ bias,  // [4096]
                            float*  __restrict__ Cst,         // [128,1024] in/out
                            __bf16* __restrict__ Hout,        // [128,1024]
                            __bf16* __restrict__ Yout)        // [128,1024] or null
{
    const int wave = blockIdx.x * (blockDim.x >> 5) + (threadIdx.x >> 5);
    const int mt = wave & 7;           // 8 batch tiles
    const int nt = wave >> 3;          // 64 hidden tiles
    const int m0 = mt << 4, n0 = nt << 4;

    v8f acc[4];
#pragma unroll
    for (int g = 0; g < 4; ++g) acc[g] = zero_v8f();

    // recurrent contribution
    for (int k0 = 0; k0 < 1024; k0 += 32) {
        v16bf a = load_fragA(Hprev, 1024, m0, k0);
#pragma unroll
        for (int g = 0; g < 4; ++g) {
            v16bf bf = load_fragB(Whh, 1024, g * 1024 + n0, k0);
            acc[g] = __builtin_amdgcn_wmma_f32_16x16x32_bf16(
                false, a, false, bf, (short)0, acc[g], false, false);
        }
    }
    // input contribution (decoder path; encoder uses precomputed Gpre)
    if (Xin) {
        for (int k0 = 0; k0 < 1024; k0 += 32) {
            v16bf a = load_fragA(Xin, 1024, m0, k0);
#pragma unroll
            for (int g = 0; g < 4; ++g) {
                v16bf bf = load_fragB(Wih, 1024, g * 1024 + n0, k0);
                acc[g] = __builtin_amdgcn_wmma_f32_16x16x32_bf16(
                    false, a, false, bf, (short)0, acc[g], false, false);
            }
        }
    }

    // fused LSTM cell epilogue (C/D tile layout: row=m0+v(+8), col=n0+lane&15)
    const int lane = threadIdx.x & 31;
    const int col  = n0 + (lane & 15);
    const int r0   = m0 + ((lane >> 4) << 3);
    const float bi = bias[col], bf_ = bias[1024 + col];
    const float bg = bias[2048 + col], bo = bias[3072 + col];
#pragma unroll
    for (int v = 0; v < 8; ++v) {
        const int b = r0 + v;
        float gi = acc[0][v] + bi;
        float gf = acc[1][v] + bf_;
        float gg = acc[2][v] + bg;
        float go = acc[3][v] + bo;
        if (Gpre) {
            const float* gp = Gpre + (size_t)b * 4096;
            gi += gp[col]; gf += gp[1024 + col];
            gg += gp[2048 + col]; go += gp[3072 + col];
        }
        const size_t idx = (size_t)b * 1024 + col;
        float cp = Cst[idx];
        float cn = sigmoidf_(gf) * cp + sigmoidf_(gi) * tanhf(gg);
        float hn = sigmoidf_(go) * tanhf(cn);
        Cst[idx] = cn;
        __bf16 hb = (__bf16)hn;
        Hout[idx] = hb;
        if (Yout) Yout[idx] = hb;
    }
}

// ---------------------------------------------------------------------------
// Attention: scores[b,s] = enc[s,b,:] . q[b,:]   (one wave per (b,s))
// ---------------------------------------------------------------------------
__global__ void k_attn_scores(const __bf16* __restrict__ enc,  // [S,B,D]
                              const __bf16* __restrict__ q,    // [B,D]
                              float* __restrict__ sc)          // [B,S]
{
    const int wave = blockIdx.x * (blockDim.x >> 5) + (threadIdx.x >> 5);
    const int b = wave >> 9, s = wave & 511;
    const int lane = threadIdx.x & 31;
    const __bf16* er = enc + ((size_t)s * 128 + b) * 1024;
    const __bf16* qr = q + (size_t)b * 1024;
    float sum = 0.f;
#pragma unroll
    for (int i = 0; i < 4; ++i) {
        const int d = (i * 32 + lane) * 8;
        v8bf e  = *(const v8bf*)(er + d);
        v8bf qq = *(const v8bf*)(qr + d);
#pragma unroll
        for (int j = 0; j < 8; ++j) sum += (float)e[j] * (float)qq[j];
    }
    sum = warp_red_sum(sum);
    if (lane == 0) sc[(size_t)b * 512 + s] = sum;
}

// softmax over S=512 per batch row; one block (256 thr) per b
__global__ void k_softmax(const float* __restrict__ sc, __bf16* __restrict__ pr)
{
    const int b = blockIdx.x, tid = threadIdx.x;
    const int lane = tid & 31, wid = tid >> 5;
    __shared__ float sred[8];
    float v0 = sc[(size_t)b * 512 + tid];
    float v1 = sc[(size_t)b * 512 + 256 + tid];
    float m = warp_red_max(fmaxf(v0, v1));
    if (lane == 0) sred[wid] = m;
    __syncthreads();
    float M = sred[0];
#pragma unroll
    for (int i = 1; i < 8; ++i) M = fmaxf(M, sred[i]);
    float e0 = __expf(v0 - M), e1 = __expf(v1 - M);
    float s = warp_red_sum(e0 + e1);
    __syncthreads();
    if (lane == 0) sred[wid] = s;
    __syncthreads();
    float S = 0.f;
#pragma unroll
    for (int i = 0; i < 8; ++i) S += sred[i];
    const float inv = 1.0f / S;
    pr[(size_t)b * 512 + tid]       = (__bf16)(e0 * inv);
    pr[(size_t)b * 512 + 256 + tid] = (__bf16)(e1 * inv);
}

// context[b,d] = sum_s probs[b,s] * enc[s,b,d]   (thread per (b,d))
__global__ void k_context(const __bf16* __restrict__ pr,
                          const __bf16* __restrict__ enc,
                          __bf16* __restrict__ ctx)
{
    const int idx = blockIdx.x * blockDim.x + threadIdx.x;   // 131072
    const int b = idx >> 10, d = idx & 1023;
    float sum = 0.f;
    for (int s = 0; s < 512; ++s)
        sum += (float)pr[(size_t)b * 512 + s] *
               (float)enc[((size_t)s * 128 + b) * 1024 + d];
    ctx[(size_t)b * 1024 + d] = (__bf16)sum;
}

// pred[b,t,c] = h[b,:] . Wfc[c,:] + bfc[c]   (one wave per (b,c))
__global__ void k_fc(const __bf16* __restrict__ h,     // [128,1024]
                     const __bf16* __restrict__ Wfc,   // [7,1024]
                     const float* __restrict__ bfc,    // [7]
                     float* __restrict__ out, int t)   // [B,96,7]
{
    const int wave = blockIdx.x * (blockDim.x >> 5) + (threadIdx.x >> 5);
    if (wave >= 128 * 7) return;
    const int b = wave / 7, c = wave % 7;
    const int lane = threadIdx.x & 31;
    const __bf16* hr = h + (size_t)b * 1024;
    const __bf16* wr = Wfc + (size_t)c * 1024;
    float sum = 0.f;
#pragma unroll
    for (int i = 0; i < 4; ++i) {
        const int d = (i * 32 + lane) * 8;
        v8bf hv = *(const v8bf*)(hr + d);
        v8bf wv = *(const v8bf*)(wr + d);
#pragma unroll
        for (int j = 0; j < 8; ++j) sum += (float)hv[j] * (float)wv[j];
    }
    sum = warp_red_sum(sum);
    if (lane == 0) out[((size_t)b * 96 + t) * 7 + c] = sum + bfc[c];
}

// ---------------------------------------------------------------------------
// Host orchestration
// ---------------------------------------------------------------------------
extern "C" void kernel_launch(void* const* d_in, const int* in_sizes, int n_in,
                              void* d_out, int out_size, void* d_ws, size_t ws_size,
                              hipStream_t stream)
{
    (void)in_sizes; (void)n_in; (void)out_size; (void)ws_size;

    const float* x       = (const float*)d_in[0];
    const float* x_mark  = (const float*)d_in[1];
    const float* conv_w  = (const float*)d_in[2];
    const float* temp_w  = (const float*)d_in[3];
    const float* enc_Wih = (const float*)d_in[4];
    const float* enc_Whh = (const float*)d_in[5];
    const float* enc_b   = (const float*)d_in[6];
    const float* dec_Wih = (const float*)d_in[7];
    const float* dec_Whh = (const float*)d_in[8];
    const float* dec_b   = (const float*)d_in[9];
    const float* W_fc    = (const float*)d_in[10];
    const float* b_fc    = (const float*)d_in[11];
    float* out = (float*)d_out;

    const size_t W4D  = (size_t)4096 * 1024;    // elems per layer weight matrix
    const size_t BD   = (size_t)128 * 1024;     // elems in one [B,D] activation
    const size_t SBD  = (size_t)512 * BD;       // elems in one [S,B,D] sequence
    const size_t MB4N = (size_t)512 * 128 * 4096;

    // workspace carve (256B aligned)
    char* base = (char*)d_ws;
    size_t off = 0;
    auto carve = [&](size_t bytes) -> void* {
        void* r = base + off;
        off += (bytes + 255) & ~(size_t)255;
        return r;
    };
    __bf16* wWihE  = (__bf16*)carve(2 * W4D * 2);
    __bf16* wWhhE  = (__bf16*)carve(2 * W4D * 2);
    __bf16* wWihD  = (__bf16*)carve(2 * W4D * 2);
    __bf16* wWhhD  = (__bf16*)carve(2 * W4D * 2);
    __bf16* wFc    = (__bf16*)carve((size_t)7 * 1024 * 2);
    __bf16* emb    = (__bf16*)carve(SBD * 2);
    __bf16* ys1    = (__bf16*)carve(SBD * 2);
    __bf16* encOut = (__bf16*)carve(SBD * 2);
    float*  G      = (float*) carve(MB4N * 4);          // [S*B, 4096] fp32
    __bf16* hbBase = (__bf16*)carve(4 * BD * 2);        // 2 layers x 2 ping-pong
    float*  cb     = (float*) carve(2 * BD * 4);        // 2 layers cell state
    float*  scores = (float*) carve((size_t)128 * 512 * 4);
    __bf16* probs  = (__bf16*)carve((size_t)128 * 512 * 2);
    __bf16* ctx    = (__bf16*)carve(BD * 2);
    auto hb = [&](int l, int slot) { return hbBase + ((size_t)(l * 2 + slot)) * BD; };

    // ---- weight conversion fp32 -> bf16 ----
    k_f32_to_bf16<<<4096, 256, 0, stream>>>(enc_Wih, wWihE, (int)(2 * W4D));
    k_f32_to_bf16<<<4096, 256, 0, stream>>>(enc_Whh, wWhhE, (int)(2 * W4D));
    k_f32_to_bf16<<<4096, 256, 0, stream>>>(dec_Wih, wWihD, (int)(2 * W4D));
    k_f32_to_bf16<<<4096, 256, 0, stream>>>(dec_Whh, wWhhD, (int)(2 * W4D));
    k_f32_to_bf16<<<28, 256, 0, stream>>>(W_fc, wFc, 7 * 1024);

    // ---- zero h0/c0 ----
    k_zero_u32<<<512, 256, 0, stream>>>((unsigned int*)hbBase, (int)(4 * BD * 2 / 4));
    k_zero_u32<<<512, 256, 0, stream>>>((unsigned int*)cb, (int)(2 * BD));

    // ---- embedding (time-major bf16) ----
    k_embed<<<512 * 128, 256, 0, stream>>>(x, x_mark, conv_w, temp_w, emb);

    // ---- encoder: 2 stacked LSTM layers ----
    for (int l = 0; l < 2; ++l) {
        const __bf16* Xseq = (l == 0) ? emb : ys1;
        __bf16* Yseq = (l == 0) ? ys1 : encOut;
        // batched input-gate GEMM: M = 65536 -> 4096 m-tiles
        k_gemm_gates<<<4096 * 16, 128, 0, stream>>>(Xseq, wWihE + l * W4D, G, 4096);
        for (int t = 0; t < 512; ++t) {
            k_lstm_step<<<128, 128, 0, stream>>>(
                nullptr, wWihE + l * W4D,
                hb(l, t & 1), wWhhE + l * W4D,
                G + (size_t)t * 128 * 4096, enc_b + l * 4096,
                cb + (size_t)l * BD, hb(l, (t + 1) & 1),
                Yseq + (size_t)t * BD);
        }
    }

    // ---- decoder: attention + 2 LSTM cells + FC, 96 steps ----
    for (int j = 0; j < 96; ++j) {
        const int p = j & 1, q = (j + 1) & 1;
        k_attn_scores<<<8192, 256, 0, stream>>>(encOut, hb(1, p), scores);
        k_softmax<<<128, 256, 0, stream>>>(scores, probs);
        k_context<<<512, 256, 0, stream>>>(probs, encOut, ctx);
        // layer 0: input = context
        k_lstm_step<<<128, 128, 0, stream>>>(
            ctx, wWihD, hb(0, p), wWhhD,
            nullptr, dec_b, cb, hb(0, q), nullptr);
        // layer 1: input = new h0
        k_lstm_step<<<128, 128, 0, stream>>>(
            hb(0, q), wWihD + W4D, hb(1, p), wWhhD + W4D,
            nullptr, dec_b + 4096, cb + BD, hb(1, q), nullptr);
        k_fc<<<112, 256, 0, stream>>>(hb(1, q), wFc, b_fc, out, j);
    }
}
